// MiniTransformerSteinshark_721554505803
// MI455X (gfx1250) — matrix operand order repaired and checked by reading, not compiled, not run on hardware
//
#include <hip/hip_runtime.h>
#include <hip/hip_bf16.h>
#include <math.h>

// Vector types for WMMA operands (CDNA5 wave32)
typedef __attribute__((ext_vector_type(16))) _Float16 v16h;
typedef __attribute__((ext_vector_type(8)))  _Float16 v8h;
typedef __attribute__((ext_vector_type(8)))  float    v8f;
// Matches the async-LDS builtin's parameter type: int __vector_size__(16)
typedef int v4i_vs __attribute__((vector_size(16)));
#define ATHENA_AS1 __attribute__((address_space(1)))
#define ATHENA_AS3 __attribute__((address_space(3)))

#if __has_builtin(__builtin_amdgcn_global_load_async_to_lds_b128)
#define ATHENA_ASYNC_LDS 1
#else
#define ATHENA_ASYNC_LDS 0
#endif

static __device__ __forceinline__ v16h ldb16(const _Float16* p) {
  v8h lo = *(const v8h*)p;
  v8h hi = *(const v8h*)(p + 8);
  return __builtin_shufflevector(lo, hi, 0, 1, 2, 3, 4, 5, 6, 7,
                                 8, 9, 10, 11, 12, 13, 14, 15);
}

// ---------------------------------------------------------------------------
// fp32 -> fp16 conversion (grid-stride)
// ---------------------------------------------------------------------------
__global__ void k_cvt_f16(const float* __restrict__ in, _Float16* __restrict__ out,
                          long long n) {
  long long i = (long long)blockIdx.x * blockDim.x + threadIdx.x;
  long long stride = (long long)gridDim.x * blockDim.x;
  for (; i < n; i += stride) out[i] = (_Float16)in[i];
}

// ---------------------------------------------------------------------------
// Embedding gather: out[b, out_row0+t, :] = sem[ids[b, col0+t], :] + pos[t, :]
// out_f16 selects output dtype (f16 for GEMM-feeding buffers, f32 for x).
// ---------------------------------------------------------------------------
__global__ void k_embed(const int* __restrict__ ids, int ids_ld, int ids_col0,
                        const float* __restrict__ sem, const float* __restrict__ pos,
                        void* __restrict__ out, long long out_bstride, int out_row0,
                        int T, int D, long long total, int out_f16) {
  long long i = (long long)blockIdx.x * blockDim.x + threadIdx.x;
  if (i >= total) return;
  int d = (int)(i % D);
  long long td = i / D;
  int t = (int)(td % T);
  int b = (int)(td / T);
  int id = ids[(long long)b * ids_ld + ids_col0 + t];
  float v = sem[(long long)id * D + d] + pos[(long long)t * D + d];
  long long off = (long long)b * out_bstride + (long long)(out_row0 + t) * D + d;
  if (out_f16) ((_Float16*)out)[off] = (_Float16)v;
  else         ((float*)out)[off] = v;
}

// ---------------------------------------------------------------------------
// WMMA GEMM workhorse:  C[M,N] = epi( A[M,K](f16) @ W[N,K](f16)^T + bias[N] )
//
// Geometry: one block (8 waves, 256 thr) computes a 32 x 512 output strip.
//  - A slab (32 rows x 512-K-chunk, f16) is staged into LDS per chunk via
//    global_load_async_to_lds_b128 (+ s_wait_asynccnt) when available;
//    rows padded by 8 halfs to spread LDS banks. 33 KB static LDS.
//  - Each wave holds 8 accumulators (2 m-tiles x 4 n-tiles). Per K-step:
//    4 LDS b128 reads give 2 shared A fragments, 8 global b128 reads give
//    4 B fragments -> 8 WMMAs. Every B fragment is reused for 2 WMMAs and
//    every A fragment for 4, halving B traffic per FLOP vs 16-row blocking.
//
// Operand layouts per CDNA5 ISA 7.12.2 (wave32):
//   A 16x32 f16 : lanes 0-15 -> row, K = {k..k+7, k+16..k+23}
//                 lanes 16-31 -> same rows, K = {k+8..k+15, k+24..k+31}
//   B 32x16 f16 : lanes 0-15 -> col, K = {k..k+7, k+16..k+23} ; lanes 16-31 -> K = {k+8..k+15, k+24..k+31}
//   C/D 16x16 f32: VGPR r -> row base+r (+8 for hi lanes), lane%16 -> column.
//
// mode 0: +bias     mode 1: +bias, exact GELU     mode 2: +bias, += C (f32)
// Requires: M % 32 == 0, N % 512 == 0, K % 512 == 0 (all shapes here comply).
// ---------------------------------------------------------------------------
__global__ __launch_bounds__(256) void k_gemm_wmma(
    const _Float16* __restrict__ A, int lda,
    const _Float16* __restrict__ W, int ldw,
    const float* __restrict__ bias,
    void* __restrict__ Cout, int ldc,
    int M, int N, int K, int mode, int out_f16)
{
  const int KC  = 512;                        // K staging chunk
  const int ldl = KC + 8;                     // padded LDS row stride (halfs)
  __shared__ _Float16 lsA[32 * (512 + 8)];    // 33 KB
  const int nblks = N >> 9;                   // N / 512
  const int m0  = (blockIdx.x / nblks) << 5;  // 32 rows per block
  const int nb0 = (blockIdx.x % nblks) << 9;
  const int wave = threadIdx.x >> 5;
  const int lane = threadIdx.x & 31;
  const int half = lane >> 4;
  const int l16  = lane & 15;
  const int n0   = nb0 + (wave << 6);

  const _Float16* Wp[4];
#pragma unroll
  for (int j = 0; j < 4; ++j)
    Wp[j] = W + (long long)(n0 + j * 16 + l16) * ldw;
  const _Float16* lA0 = &lsA[l16 * ldl];          // m-tile 0 rows
  const _Float16* lA1 = &lsA[(16 + l16) * ldl];   // m-tile 1 rows

  v8f cc[2][4] = {{{}, {}, {}, {}}, {{}, {}, {}, {}}};

  for (int kk = 0; kk < K; kk += KC) {
    // ---- stage A[m0:m0+32, kk:kk+512] into LDS (16B chunks, 256 threads) ----
    for (int c = threadIdx.x; c < 32 * 64; c += 256) {   // 64 chunks per row
      const int row = c >> 6;
      const int kc  = (c & 63) << 3;
      const _Float16* gsrc = A + (long long)(m0 + row) * lda + kk + kc;
      _Float16* ldst = &lsA[row * ldl + kc];
#if ATHENA_ASYNC_LDS
      __builtin_amdgcn_global_load_async_to_lds_b128(
          (ATHENA_AS1 v4i_vs*)(_Float16*)gsrc,
          (ATHENA_AS3 v4i_vs*)ldst, 0, 0);
#else
      *(v8h*)ldst = *(const v8h*)gsrc;
#endif
    }
#if ATHENA_ASYNC_LDS
#if __has_builtin(__builtin_amdgcn_s_wait_asynccnt)
    __builtin_amdgcn_s_wait_asynccnt(0);
#else
    asm volatile("s_wait_asynccnt 0" ::: "memory");
#endif
#endif
    __syncthreads();

    for (int k = 0; k < KC; k += 32) {
      const int ka = k + half * 8;             // A K-group (LDS offset)
      const int kb = kk + k + half * 8;        // B K-group (global offset)
      __builtin_prefetch(Wp[0] + kb + 512, 0, 1);  // -> global_prefetch_b8
      v8h a00 = *(const v8h*)(lA0 + ka);
      v8h a01 = *(const v8h*)(lA0 + ka + 16);
      v16h a0 = __builtin_shufflevector(a00, a01, 0, 1, 2, 3, 4, 5, 6, 7,
                                        8, 9, 10, 11, 12, 13, 14, 15);
      v8h a10 = *(const v8h*)(lA1 + ka);
      v8h a11 = *(const v8h*)(lA1 + ka + 16);
      v16h a1 = __builtin_shufflevector(a10, a11, 0, 1, 2, 3, 4, 5, 6, 7,
                                        8, 9, 10, 11, 12, 13, 14, 15);
#pragma unroll
      for (int j = 0; j < 4; ++j) {
        v16h b = __builtin_shufflevector(*(const v8h*)(Wp[j] + kb), *(const v8h*)(Wp[j] + kb + 16),
                                         0, 1, 2, 3, 4, 5, 6, 7, 8, 9, 10, 11, 12, 13, 14, 15);
        cc[0][j] = __builtin_amdgcn_wmma_f32_16x16x32_f16(
            false, a0, false, b, (short)0, cc[0][j], false, false);
        cc[1][j] = __builtin_amdgcn_wmma_f32_16x16x32_f16(
            false, a1, false, b, (short)0, cc[1][j], false, false);
      }
    }
    __syncthreads();   // protect LDS before next chunk overwrites
  }

  // ---- fused epilogue: bias / GELU / residual, f16 or f32 store ----
#pragma unroll
  for (int mt = 0; mt < 2; ++mt) {
#pragma unroll
    for (int j = 0; j < 4; ++j) {
      const int col = n0 + j * 16 + l16;
      const float bv = bias[col];
#pragma unroll
      for (int r = 0; r < 8; ++r) {
        const int row = m0 + mt * 16 + r + half * 8;
        float v = cc[mt][j][r] + bv;
        if (mode == 1) v = 0.5f * v * (1.f + erff(v * 0.70710678118654752f));
        const long long off = (long long)row * ldc + col;
        if (out_f16) {
          ((_Float16*)Cout)[off] = (_Float16)v;
        } else {
          float* Cf = (float*)Cout;
          if (mode == 2) v += Cf[off];
          Cf[off] = v;
        }
      }
    }
  }
}

// ---------------------------------------------------------------------------
// Fused attention (flash-style, one q-row per thread, dk = 32 in registers).
// f16 Q/K/V in, fp32 math, f16 O out. One block per (batch, head).
// q_bstride = 0 broadcasts the summarizers' shared learned queries.
// ---------------------------------------------------------------------------
__global__ void k_attn(const _Float16* __restrict__ Q, long long q_bstride, int ldq, int Tq,
                       const _Float16* __restrict__ Km, const _Float16* __restrict__ Vm,
                       long long kv_bstride, int ldkv, int Tk,
                       _Float16* __restrict__ O, long long o_bstride, int ldo,
                       int nheads, int causal)
{
  const int b = blockIdx.x / nheads;
  const int h = blockIdx.x % nheads;
  const _Float16* Qb = Q + (long long)b * q_bstride + h * 32;
  const _Float16* Kb = Km + (long long)b * kv_bstride + h * 32;
  const _Float16* Vb = Vm + (long long)b * kv_bstride + h * 32;
  _Float16* Ob = O + (long long)b * o_bstride + h * 32;
  const float scale = 0.17677669529663687f;  // 1/sqrt(32)

  for (int t = threadIdx.x; t < Tq; t += blockDim.x) {
    const _Float16* q = Qb + (long long)t * ldq;
    float qv[32];
#pragma unroll
    for (int d = 0; d < 32; ++d) qv[d] = (float)q[d];
    float m = -INFINITY, ssum = 0.f;
    float acc[32];
#pragma unroll
    for (int d = 0; d < 32; ++d) acc[d] = 0.f;
    const int kmax = causal ? (t + 1) : Tk;
    for (int j = 0; j < kmax; ++j) {
      const _Float16* kr = Kb + (long long)j * ldkv;
      float s = 0.f;
#pragma unroll
      for (int d = 0; d < 32; ++d) s += qv[d] * (float)kr[d];
      s *= scale;
      const float mn = fmaxf(m, s);
      const float cold = __expf(m - mn);
      const float p = __expf(s - mn);
      ssum = ssum * cold + p;
      const _Float16* vr = Vb + (long long)j * ldkv;
#pragma unroll
      for (int d = 0; d < 32; ++d) acc[d] = acc[d] * cold + p * (float)vr[d];
      m = mn;
    }
    const float inv = 1.f / ssum;
    _Float16* o = Ob + (long long)t * ldo;
#pragma unroll
    for (int d = 0; d < 32; ++d) o[d] = (_Float16)(acc[d] * inv);
  }
}

// ---------------------------------------------------------------------------
// LayerNorm (one row per block, fp32 in; f16 or f32 out).
// Generic segment mapping handles scatter into x and gathering head rows.
// ---------------------------------------------------------------------------
__global__ void k_ln(const float* __restrict__ in, const float* __restrict__ g,
                     const float* __restrict__ bb, void* __restrict__ out, int D,
                     int seg_rows, long long in_seg_stride, int in_row0,
                     long long out_seg_stride, int out_row0, int out_f16)
{
  const int r = blockIdx.x;
  const int seg = r / seg_rows, rr = r % seg_rows;
  const float* x = in + (long long)seg * in_seg_stride + (long long)(in_row0 + rr) * D;
  const long long yoff = (long long)seg * out_seg_stride + (long long)(out_row0 + rr) * D;
  __shared__ float red0[8], red1[8];
  float s = 0.f, s2 = 0.f;
  for (int i = threadIdx.x; i < D; i += blockDim.x) {
    float v = x[i]; s += v; s2 += v * v;
  }
  for (int off = 16; off > 0; off >>= 1) {   // wave32 reduce
    s  += __shfl_down(s,  off, 32);
    s2 += __shfl_down(s2, off, 32);
  }
  const int wave = threadIdx.x >> 5, lane = threadIdx.x & 31;
  if (lane == 0) { red0[wave] = s; red1[wave] = s2; }
  __syncthreads();
  if (threadIdx.x == 0) {
    float a = 0.f, c = 0.f;
    const int nw = blockDim.x >> 5;
    for (int w = 0; w < nw; ++w) { a += red0[w]; c += red1[w]; }
    red0[0] = a; red1[0] = c;
  }
  __syncthreads();
  const float mean = red0[0] / D;
  const float var  = red1[0] / D - mean * mean;
  const float rstd = rsqrtf(var + 1e-5f);
  for (int i = threadIdx.x; i < D; i += blockDim.x) {
    float v = (x[i] - mean) * rstd * g[i] + bb[i];
    if (out_f16) ((_Float16*)out)[yoff + i] = (_Float16)v;
    else         ((float*)out)[yoff + i] = v;
  }
}

// ---------------------------------------------------------------------------
// Second tuple output: tgt[b, t] = target_ids[b, col0+t]  (int32 bit patterns)
// ---------------------------------------------------------------------------
__global__ void k_tgt(const int* __restrict__ tids, int* __restrict__ out,
                      int T_in, int col0, int core, int n) {
  int i = blockIdx.x * blockDim.x + threadIdx.x;
  if (i >= n) return;
  int b = i / core, t = i % core;
  out[i] = tids[(long long)b * T_in + col0 + t];
}

// ---------------------------------------------------------------------------
extern "C" void kernel_launch(void* const* d_in, const int* in_sizes, int n_in,
                              void* d_out, int out_size, void* d_ws, size_t ws_size,
                              hipStream_t stream)
{
  (void)in_sizes; (void)n_in; (void)out_size; (void)ws_size;
  enum { B = 16, TIN = 1024, D = 512, NH = 16, L = 16, F = 1024, V = 32768,
         C1 = 512, C2 = 256, CORE = 256, LG = 64, MD = 64, T = 384 };

  const int*   input_ids  = (const int*)d_in[0];
  const int*   target_ids = (const int*)d_in[1];
  const float* emb_sem = (const float*)d_in[2];
  const float* emb_pos = (const float*)d_in[3];
  const float* lg_sem  = (const float*)d_in[4];
  const float* lg_pos  = (const float*)d_in[5];
  const float* lg_inw  = (const float*)d_in[6];
  const float* lg_inb  = (const float*)d_in[7];
  const float* lg_outw = (const float*)d_in[8];
  const float* lg_outb = (const float*)d_in[9];
  const float* lg_tok  = (const float*)d_in[10];
  const float* lg_lng  = (const float*)d_in[11];
  const float* lg_lnb  = (const float*)d_in[12];
  const float* md_sem  = (const float*)d_in[13];
  const float* md_pos  = (const float*)d_in[14];
  const float* md_inw  = (const float*)d_in[15];
  const float* md_inb  = (const float*)d_in[16];
  const float* md_outw = (const float*)d_in[17];
  const float* md_outb = (const float*)d_in[18];
  const float* md_tok  = (const float*)d_in[19];
  const float* md_lng  = (const float*)d_in[20];
  const float* md_lnb  = (const float*)d_in[21];
  const float* qkv_w   = (const float*)d_in[22];
  const float* qkv_b   = (const float*)d_in[23];
  const float* wo_w    = (const float*)d_in[24];
  const float* wo_b    = (const float*)d_in[25];
  const float* ln1_g   = (const float*)d_in[26];
  const float* ln1_b   = (const float*)d_in[27];
  const float* ln2_g   = (const float*)d_in[28];
  const float* ln2_b   = (const float*)d_in[29];
  const float* ff1_w   = (const float*)d_in[30];
  const float* ff1_b   = (const float*)d_in[31];
  const float* ff2_w   = (const float*)d_in[32];
  const float* ff2_b   = (const float*)d_in[33];
  const float* hd_lng  = (const float*)d_in[34];
  const float* hd_lnb  = (const float*)d_in[35];
  const float* hd_w    = (const float*)d_in[36];
  const float* hd_b    = (const float*)d_in[37];

  // ---- workspace carve: f16 weights (~105 MB, L2-resident) + activations ----
  char* p = (char*)d_ws;
  auto carve = [&](long long bytes) -> void* {
    void* r = (void*)p; p += (bytes + 255) & ~255LL; return r;
  };
  _Float16* w_lg_in  = (_Float16*)carve((long long)3 * D * D * 2);
  _Float16* w_lg_out = (_Float16*)carve((long long)D * D * 2);
  _Float16* w_md_in  = (_Float16*)carve((long long)3 * D * D * 2);
  _Float16* w_md_out = (_Float16*)carve((long long)D * D * 2);
  _Float16* w_qkv    = (_Float16*)carve((long long)L * 3 * D * D * 2);
  _Float16* w_wo     = (_Float16*)carve((long long)L * D * D * 2);
  _Float16* w_ff1    = (_Float16*)carve((long long)L * F * D * 2);
  _Float16* w_ff2    = (_Float16*)carve((long long)L * D * F * 2);
  _Float16* w_hd     = (_Float16*)carve((long long)V * D * 2);
  _Float16* tok16a   = (_Float16*)carve((long long)LG * D * 2);
  _Float16* tok16b   = (_Float16*)carve((long long)MD * D * 2);
  float*    x        = (float*)carve((long long)B * T * D * 4);     // residual (fp32)
  _Float16* hbuf16   = (_Float16*)carve((long long)B * T * D * 2);  // LN out
  _Float16* qkvbuf16 = (_Float16*)carve((long long)B * T * 3 * D * 2);
  _Float16* obuf16   = (_Float16*)carve((long long)B * T * D * 2);
  _Float16* fbuf16   = (_Float16*)carve((long long)B * T * F * 2);
  _Float16* sumx16   = (_Float16*)carve((long long)B * C1 * D * 2);
  _Float16* kvbuf16  = (_Float16*)carve((long long)B * C1 * 2 * D * 2);
  _Float16* qsum16   = (_Float16*)carve((long long)LG * D * 2);
  _Float16* sumo16   = (_Float16*)carve((long long)B * LG * D * 2);
  float*    sumpj    = (float*)carve((long long)B * LG * D * 4);
  _Float16* hlast16  = (_Float16*)carve((long long)B * CORE * D * 2);

  auto cvt = [&](const float* src, _Float16* dst, long long n) {
    k_cvt_f16<<<1024, 256, 0, stream>>>(src, dst, n);
  };
  auto gemm = [&](const _Float16* A, int lda, const _Float16* Wt, int ldw,
                  const float* bias, void* Cp, int ldc,
                  int M, int N, int K, int mode, int out_f16) {
    int blocks = (M >> 5) * (N >> 9);   // 32 x 512 strip per block
    k_gemm_wmma<<<blocks, 256, 0, stream>>>(A, lda, Wt, ldw, bias, Cp, ldc,
                                            M, N, K, mode, out_f16);
  };
  auto ln = [&](const float* in, const float* g, const float* bb, void* out, int rows,
                int seg_rows, long long in_ss, int in_r0, long long out_ss, int out_r0,
                int out_f16) {
    k_ln<<<rows, 128, 0, stream>>>(in, g, bb, out, D, seg_rows, in_ss, in_r0,
                                   out_ss, out_r0, out_f16);
  };

  // ---- convert all weights + learned tokens fp32 -> f16 ----
  cvt(lg_inw,  w_lg_in,  (long long)3 * D * D);
  cvt(lg_outw, w_lg_out, (long long)D * D);
  cvt(md_inw,  w_md_in,  (long long)3 * D * D);
  cvt(md_outw, w_md_out, (long long)D * D);
  cvt(qkv_w,   w_qkv,    (long long)L * 3 * D * D);
  cvt(wo_w,    w_wo,     (long long)L * D * D);
  cvt(ff1_w,   w_ff1,    (long long)L * F * D);
  cvt(ff2_w,   w_ff2,    (long long)L * D * F);
  cvt(hd_w,    w_hd,     (long long)V * D);
  cvt(lg_tok,  tok16a,   (long long)LG * D);
  cvt(md_tok,  tok16b,   (long long)MD * D);

  // ---- summarizers (lg then md), outputs scattered into x rows [0..127] ----
  struct Summ {
    const float* sem; const float* pos; const _Float16* win; const float* binb;
    const _Float16* wout; const float* boutb; const _Float16* tok;
    const float* lng; const float* lnb; int C; int col0; int xrow0;
  };
  const Summ SS[2] = {
    { lg_sem, lg_pos, w_lg_in, lg_inb, w_lg_out, lg_outb, tok16a, lg_lng, lg_lnb, C1, 0,  0  },
    { md_sem, md_pos, w_md_in, md_inb, w_md_out, md_outb, tok16b, md_lng, md_lnb, C2, C1, LG }
  };
  for (int s = 0; s < 2; ++s) {
    const Summ& S = SS[s];
    const long long total = (long long)B * S.C * D;
    k_embed<<<(int)((total + 255) / 256), 256, 0, stream>>>(
        input_ids, TIN, S.col0, S.sem, S.pos, sumx16, (long long)S.C * D, 0,
        S.C, D, total, /*f16*/1);
    // q = tok @ wq^T + bq  (batch-invariant learned queries)
    gemm(S.tok, D, S.win, D, S.binb, qsum16, D, LG, D, D, 0, 1);
    // kv = x @ [wk; wv]^T + [bk; bv]
    gemm(sumx16, D, S.win + (long long)D * D, D, S.binb + D, kvbuf16, 2 * D,
         B * S.C, 2 * D, D, 0, 1);
    // cross-attention (no mask)
    k_attn<<<B * NH, 64, 0, stream>>>(qsum16, 0LL, D, LG,
                                      kvbuf16, kvbuf16 + D,
                                      (long long)S.C * 2 * D, 2 * D, S.C,
                                      sumo16, (long long)LG * D, D, NH, 0);
    // output projection (fp32 out, LN follows) + LN scatter into x
    gemm(sumo16, D, S.wout, D, S.boutb, sumpj, D, B * LG, D, D, 0, 0);
    ln(sumpj, S.lng, S.lnb, x, B * LG, LG, (long long)LG * D, 0,
       (long long)T * D, S.xrow0, /*f32*/0);
  }

  // ---- core token embedding into x rows [128..383] (fp32 residual) ----
  {
    const long long total = (long long)B * CORE * D;
    k_embed<<<(int)((total + 255) / 256), 256, 0, stream>>>(
        input_ids, TIN, C1 + C2, emb_sem, emb_pos, x, (long long)T * D, LG + MD,
        CORE, D, total, /*f32*/0);
  }

  // ---- transformer layers (pre-LN, causal) ----
  for (int l = 0; l < L; ++l) {
    ln(x, ln1_g + l * D, ln1_b + l * D, hbuf16, B * T, T,
       (long long)T * D, 0, (long long)T * D, 0, 1);
    gemm(hbuf16, D, w_qkv + (long long)l * 3 * D * D, D, qkv_b + (long long)l * 3 * D,
         qkvbuf16, 3 * D, B * T, 3 * D, D, 0, 1);
    k_attn<<<B * NH, 128, 0, stream>>>(qkvbuf16, (long long)T * 3 * D, 3 * D, T,
                                       qkvbuf16 + D, qkvbuf16 + 2 * D,
                                       (long long)T * 3 * D, 3 * D, T,
                                       obuf16, (long long)T * D, D, NH, 1);
    gemm(obuf16, D, w_wo + (long long)l * D * D, D, wo_b + l * D, x, D,
         B * T, D, D, 2, 0);
    ln(x, ln2_g + l * D, ln2_b + l * D, hbuf16, B * T, T,
       (long long)T * D, 0, (long long)T * D, 0, 1);
    gemm(hbuf16, D, w_ff1 + (long long)l * F * D, D, ff1_b + (long long)l * F,
         fbuf16, F, B * T, F, D, 1, 1);
    gemm(fbuf16, F, w_ff2 + (long long)l * D * F, F, ff2_b + l * D, x, D,
         B * T, D, F, 2, 0);
  }

  // ---- head: LN over last CORE rows -> f16, big WMMA GEMM into d_out ----
  ln(x, hd_lng, hd_lnb, hlast16, B * CORE, CORE, (long long)T * D, LG + MD,
     (long long)CORE * D, 0, 1);
  float* logits = (float*)d_out;
  gemm(hlast16, D, w_hd, D, hd_b, logits, V, B * CORE, V, D, 0, 0);

  // ---- second tuple output: tgt = target_ids[:, -CORE:] ----
  int* tgt_out = (int*)((float*)d_out + (long long)B * CORE * V);
  k_tgt<<<(B * CORE + 255) / 256, 256, 0, stream>>>(target_ids, tgt_out, TIN, C1 + C2,
                                                    CORE, B * CORE);
}
